// SlotBSDDynamicsModel_86887188398406
// MI455X (gfx1250) — matrix operand order repaired and checked by reading, not compiled, not run on hardware
//
#include <hip/hip_runtime.h>
#include <hip/hip_bf16.h>

// ---------------------------------------------------------------------------
// SlotBSDDynamicsModel for MI455X (gfx1250, wave32, WMMA bf16)
// - All B operands stored [N,K] so A and B WMMA fragments are 2x ds_load_b128.
// - GEMM: double-buffered LDS pipeline using CDNA5 async global->LDS copies
//   (ASYNCcnt + s_wait_asynccnt), one barrier per K-tile, zero guards
//   (operands padded so M%128==0, N%64==0, K%32==0 always hold).
// ---------------------------------------------------------------------------

typedef __bf16 bf16_t;
typedef __attribute__((ext_vector_type(16))) __bf16 v16bf;
typedef __attribute__((ext_vector_type(8)))  __bf16 v8bf;
typedef __attribute__((ext_vector_type(8)))  float  v8f;

#ifndef __has_builtin
#define __has_builtin(x) 0
#endif
#if __has_builtin(__builtin_amdgcn_global_load_async_to_lds_b128) && \
    __has_builtin(__builtin_amdgcn_s_wait_asynccnt)
#define HAVE_ASYNC_LDS 1
typedef int v4i_a __attribute__((vector_size(16)));
typedef __attribute__((address_space(1))) v4i_a* gptr_v4i;
typedef __attribute__((address_space(3))) v4i_a* lptr_v4i;
#define COPY16(dst, src)                                    \
  __builtin_amdgcn_global_load_async_to_lds_b128(           \
      (gptr_v4i)(uintptr_t)(src), (lptr_v4i)(uintptr_t)(dst), 0, 0)
#else
#define HAVE_ASYNC_LDS 0
#define COPY16(dst, src)                                    \
  (*reinterpret_cast<uint4*>(dst) = *reinterpret_cast<const uint4*>(src))
#endif

// Model constants
#define MB   256        // batch
#define MT   64         // time
#define NTOK (MB*MT)    // 16384 tokens
#define DM   512        // d_model
#define DS   256        // d_state
#define NL   6
#define MH   2048       // mlp hidden
#define HID  512        // decoder hidden
#define LAT  512        // C*H*W
#define AKP  32         // action K padded (18 -> 32)

__device__ __forceinline__ float gelu_f(float x) {
  return 0.5f * x * (1.0f + erff(x * 0.70710678118654752f));
}
__device__ __forceinline__ float sigmoid_f(float x) {
  return 1.0f / (1.0f + __expf(-x));
}

// ----------------------- WMMA fragment helpers (LDS) -----------------------
__device__ __forceinline__ v16bf frag_a_ld(const bf16_t* __restrict__ s, int ld) {
  const int lane = threadIdx.x & 31;
  const int row  = lane & 15;
  const int kh   = (lane >> 4) << 3;
  const v8bf lo = *reinterpret_cast<const v8bf*>(s + row * ld + kh);
  const v8bf hi = *reinterpret_cast<const v8bf*>(s + row * ld + kh + 16);
  return __builtin_shufflevector(lo, hi, 0, 1, 2, 3, 4, 5, 6, 7, 8, 9, 10, 11,
                                 12, 13, 14, 15);
}
// B operand stored [N,K] row-major.
__device__ __forceinline__ v16bf frag_b_ld(const bf16_t* __restrict__ s, int ld) {
  const int lane = threadIdx.x & 31;
  const int n  = lane & 15;
  const int k0 = (lane >> 4) << 3;
  const v8bf lo = *reinterpret_cast<const v8bf*>(s + n * ld + k0);
  const v8bf hi = *reinterpret_cast<const v8bf*>(s + n * ld + k0 + 16);
  return __builtin_shufflevector(lo, hi, 0, 1, 2, 3, 4, 5, 6, 7, 8, 9, 10, 11,
                                 12, 13, 14, 15);
}
__device__ __forceinline__ v8f wmma_bf16(v16bf a, v16bf b, v8f c) {
  return __builtin_amdgcn_wmma_f32_16x16x32_bf16(false, a, false, b, (short)0, c,
                                                 false, false);
}
// C/D mapping: VGPR r holds (M = r + (lane>=16 ? 8 : 0), N = lane&15)

// ----------------------- f32 -> bf16 (+optional transpose) -----------------
__global__ void cvt_kernel(const float* __restrict__ src, bf16_t* __restrict__ dst,
                           int rows, int cols, int do_transpose) {
  const int i = blockIdx.x * blockDim.x + threadIdx.x;
  const int total = rows * cols;
  if (i >= total) return;
  const int r = i / cols, c = i % cols;
  const float v = src[i];
  if (do_transpose) dst[(size_t)c * rows + r] = (bf16_t)v;
  else              dst[i] = (bf16_t)v;
}

// f32 -> bf16 with zero-padding of the K (fastest) dimension to dstK.
// no-transpose: src[rows,cols] -> dst[rows,dstK]
// transpose:    src[rows,cols] -> dst[cols,dstK] (dst row c holds src column c)
__global__ void cvt_pad_kernel(const float* __restrict__ src,
                               bf16_t* __restrict__ dst, int rows, int cols,
                               int dstK, int do_transpose) {
  const int i = blockIdx.x * blockDim.x + threadIdx.x;
  if (!do_transpose) {
    if (i >= rows * dstK) return;
    const int r = i / dstK, k = i % dstK;
    dst[i] = (k < cols) ? (bf16_t)src[(size_t)r * cols + k] : (bf16_t)0.0f;
  } else {
    if (i >= cols * dstK) return;
    const int c = i / dstK, k = i % dstK;
    dst[i] = (k < rows) ? (bf16_t)src[(size_t)k * cols + c] : (bf16_t)0.0f;
  }
}

// ----------------------- generic bf16 GEMM (WMMA) --------------------------
// C[M,N] = epilogue( A[M,K]bf16 @ (Bt[N,K]bf16)^T + bias + resid )
// Preconditions (guaranteed by call sites): M%128==0, N%64==0, K%32==0.
#define GTM 128
#define GTN 64
#define GTK 32
#define A_LD 40   // 80B rows: 16B aligned, conflict-free lane stride
#define B_LD 40

__global__ __launch_bounds__(256) void gemm_bf16_kernel(
    const bf16_t* __restrict__ A, const bf16_t* __restrict__ Bt,
    const float* __restrict__ bias, const float* __restrict__ resid,
    float* __restrict__ Cf, bf16_t* __restrict__ Cb,
    int M, int N, int K, int do_gelu) {
  __shared__ __align__(16) bf16_t As[2][GTM * A_LD];  // 2 x 10 KB
  __shared__ __align__(16) bf16_t Bs[2][GTN * B_LD];  // 2 x 5 KB
  const int tid  = threadIdx.x;
  const int lane = tid & 31;
  const int wave = tid >> 5;
  const int m0 = blockIdx.y * GTM;
  const int n0 = blockIdx.x * GTN;

  // Issue one K-tile's worth of async global->LDS copies (16B per lane-op).
  auto stage = [&](int buf, int k0) {
#pragma unroll
    for (int it = 0; it < 2; ++it) {
      const int chunk = it * 256 + tid;
      const int mm = chunk >> 2, h8 = (chunk & 3) * 8;
      COPY16(&As[buf][mm * A_LD + h8], A + (size_t)(m0 + mm) * K + k0 + h8);
    }
    const int nn = tid >> 2, h8 = (tid & 3) * 8;
    COPY16(&Bs[buf][nn * B_LD + h8], Bt + (size_t)(n0 + nn) * K + k0 + h8);
  };

  v8f acc[4] = {};
  const int kTiles = K / GTK;

  stage(0, 0);
  for (int kt = 0; kt < kTiles; ++kt) {
    const int cur = kt & 1;
    // Wait for this tile's copies; barrier also proves every wave finished the
    // previous tile's WMMA, so the other buffer is free to overwrite.
#if HAVE_ASYNC_LDS
    __builtin_amdgcn_s_wait_asynccnt(0);
#endif
    __syncthreads();
    if (kt + 1 < kTiles) {
      stage(1 - cur, (kt + 1) * GTK);  // overlaps with WMMA below
      if (kt + 2 < kTiles) {           // prefetch tile kt+2 into cache
        const int pk = (kt + 2) * GTK;
        __builtin_prefetch(A + (size_t)(m0 + (tid >> 1)) * K + pk + (tid & 1) * 16, 0, 1);
        __builtin_prefetch(Bt + (size_t)(n0 + (tid & 63)) * K + pk, 0, 1);
      }
    }
    const v16bf af = frag_a_ld(&As[cur][wave * 16 * A_LD], A_LD);
#pragma unroll
    for (int nb = 0; nb < 4; ++nb)
      acc[nb] = wmma_bf16(af, frag_b_ld(&Bs[cur][nb * 16 * B_LD], B_LD), acc[nb]);
  }
  // ---- epilogue (no bounds checks needed) ----
  const int rowbase = m0 + wave * 16 + ((lane >> 4) << 3);
  const int colbase = n0 + (lane & 15);
#pragma unroll
  for (int nb = 0; nb < 4; ++nb) {
    const int col = colbase + nb * 16;
    const float bv = bias ? bias[col] : 0.0f;
#pragma unroll
    for (int r = 0; r < 8; ++r) {
      const int row = rowbase + r;
      float v = acc[nb][r] + bv;
      if (resid) v += resid[(size_t)row * N + col];
      if (do_gelu) v = gelu_f(v);
      if (Cf) Cf[(size_t)row * N + col] = v;
      if (Cb) Cb[(size_t)row * N + col] = (bf16_t)v;
    }
  }
}

// ----------------------- row LayerNorm (f32 in -> bf16/f32 out) ------------
__global__ __launch_bounds__(256) void ln_kernel(
    const float* __restrict__ x, const float* __restrict__ g,
    const float* __restrict__ b, bf16_t* __restrict__ outb,
    float* __restrict__ outf, const float* __restrict__ scale, int D) {
  const int row = blockIdx.x;
  const float* xr = x + (size_t)row * D;
  const int tid = threadIdx.x;
  float s = 0.f, ss = 0.f;
  for (int i = tid; i < D; i += 256) {
    const float v = xr[i];
    s += v; ss += v * v;
  }
#pragma unroll
  for (int off = 16; off > 0; off >>= 1) {
    s  += __shfl_xor(s, off, 32);
    ss += __shfl_xor(ss, off, 32);
  }
  __shared__ float rs[8], rss[8];
  if ((tid & 31) == 0) { rs[tid >> 5] = s; rss[tid >> 5] = ss; }
  __syncthreads();
  float ts = 0.f, tss = 0.f;
#pragma unroll
  for (int i = 0; i < 8; ++i) { ts += rs[i]; tss += rss[i]; }
  const float mean = ts / (float)D;
  const float inv  = rsqrtf(tss / (float)D - mean * mean + 1e-5f);
  const float sc   = scale ? scale[0] : 1.0f;
  for (int i = tid; i < D; i += 256) {
    const float v = ((xr[i] - mean) * inv * g[i] + b[i]) * sc;
    if (outb) outb[(size_t)row * D + i] = (bf16_t)v;
    if (outf) outf[(size_t)row * D + i] = v;
  }
}

// ----------------------- spectral diagonal SSM scan ------------------------
__global__ __launch_bounds__(256) void ssm_scan_kernel(
    const float* __restrict__ u, const float* __restrict__ lam,
    const float* __restrict__ theta, bf16_t* __restrict__ ys,
    int Bn, int Tn, int Sn) {
  const int idx = blockIdx.x * blockDim.x + threadIdx.x;
  if (idx >= Bn * Sn) return;
  const int s = idx % Sn, b = idx / Sn;
  const float mag = 0.9f + 0.099f * sigmoid_f(lam[s]);
  const float th = 0.1f * theta[s];
  const float a = mag * cosf(th), bb = mag * sinf(th);
  float sr = 0.f, si = 0.f;
  const float* up = u + (size_t)b * Tn * Sn + s;
  bf16_t* yp = ys + (size_t)b * Tn * Sn + s;
  for (int t = 0; t < Tn; ++t) {
    const float ut = up[(size_t)t * Sn];
    const float sr2 = a * sr - bb * si + ut;
    si = bb * sr + a * si;
    sr = sr2;
    yp[(size_t)t * Sn] = (bf16_t)sr;
  }
}

// ----------------------- slot attention scan -------------------------------
// One workgroup (8 waves) per batch element; all weights resident in LDS
// (~234 KB of the 320 KB WGP LDS). Every B operand stored [N,K].
#define SLOT_LDS_FLOATS 14336
#define SLOT_LDS_BF16   91136
#define SLOT_LDS_BYTES  (SLOT_LDS_FLOATS * 4 + SLOT_LDS_BF16 * 2)

__global__ __launch_bounds__(256) void slot_attn_kernel(
    const float* __restrict__ g_lat,    // [B,T,512] (C,H,W flattened)
    const float* __restrict__ g_prev,   // [B,8,64]
    const float* __restrict__ g_projw,  // [8,64]
    const float* __restrict__ g_projb,  // [64]
    const float* __restrict__ g_pos,    // [64,64]
    const float* __restrict__ g_ang, const float* __restrict__ g_anb,
    const bf16_t* __restrict__ g_wq, const bf16_t* __restrict__ g_wk,
    const bf16_t* __restrict__ g_wv,                     // [out=64, in=64]
    const bf16_t* __restrict__ g_wih, const bf16_t* __restrict__ g_whh,  // [192,64]
    const float* __restrict__ g_bih, const float* __restrict__ g_bhh,
    const float* __restrict__ g_sng, const float* __restrict__ g_snb,
    const bf16_t* __restrict__ g_sw1T, const float* __restrict__ g_sb1,  // [256,64]
    const bf16_t* __restrict__ g_sw2T, const float* __restrict__ g_sb2,  // [64,256]
    float* __restrict__ all_slots,   // [B,T,512]
    float* __restrict__ final_slots, // [B,512] -> d_out tail
    int Tn) {
  extern __shared__ char smem_raw[];
  float* fp = reinterpret_cast<float*>(smem_raw);
  float* posf   = fp;            fp += 4096;
  float* tokf   = fp;            fp += 4096;
  float* projw  = fp;            fp += 512;
  float* pjbf   = fp;            fp += 64;
  float* latf   = fp;            fp += 512;
  float* attnf  = fp;            fp += 512;   // [8,64] logits
  float* gxf    = fp;            fp += 1536;  // [8,192]
  float* ghf    = fp;            fp += 1536;
  float* slotsf = fp;            fp += 512;   // [8,64]
  float* angf   = fp;            fp += 64;
  float* anbf   = fp;            fp += 64;
  float* sngf   = fp;            fp += 64;
  float* snbf   = fp;            fp += 64;
  float* bihf   = fp;            fp += 192;
  float* bhhf   = fp;            fp += 192;
  float* sb1f   = fp;            fp += 256;
  float* sb2f   = fp;            fp += 64;
  bf16_t* bp = reinterpret_cast<bf16_t*>(fp);
  bf16_t* tok    = bp;  bp += 4096;   // LN'd tokens [n=64, d=64] (A operand)
  bf16_t* kk     = bp;  bp += 4096;   // k natural [token, d]  (B of q@k^T)
  bf16_t* vT     = bp;  bp += 4096;   // v transposed [d, token] (B of attn@v)
  bf16_t* slotsb = bp;  bp += 1024;   // [16,64] rows 8..15 zero
  bf16_t* qb     = bp;  bp += 1024;
  bf16_t* attnb  = bp;  bp += 1024;
  bf16_t* updb   = bp;  bp += 1024;
  bf16_t* hbs    = bp;  bp += 1024;
  bf16_t* hidb   = bp;  bp += 4096;   // [16,256]
  bf16_t* wqs    = bp;  bp += 4096;   // [64,64]  (N=out,K=in)
  bf16_t* wks    = bp;  bp += 4096;
  bf16_t* wvs    = bp;  bp += 4096;
  bf16_t* wihs   = bp;  bp += 12288;  // [192,64]
  bf16_t* whhs   = bp;  bp += 12288;
  bf16_t* sw1s   = bp;  bp += 16384;  // [256,64]
  bf16_t* sw2s   = bp;  bp += 16384;  // [64,256]

  const int tid  = threadIdx.x;
  const int lane = tid & 31;
  const int wave = tid >> 5;
  const int b    = blockIdx.x;

  // ---- stage constants & weights into LDS ----
  for (int i = tid; i < 4096; i += 256) posf[i]  = g_pos[i];
  for (int i = tid; i < 512;  i += 256) projw[i] = g_projw[i];
  for (int i = tid; i < 64;   i += 256) pjbf[i]  = g_projb[i];
  for (int i = tid; i < 64;   i += 256) { angf[i] = g_ang[i]; anbf[i] = g_anb[i];
                                          sngf[i] = g_sng[i]; snbf[i] = g_snb[i];
                                          sb2f[i] = g_sb2[i]; }
  for (int i = tid; i < 192;  i += 256) { bihf[i] = g_bih[i]; bhhf[i] = g_bhh[i]; }
  for (int i = tid; i < 256;  i += 256) sb1f[i] = g_sb1[i];
  for (int i = tid; i < 4096; i += 256) { wqs[i] = g_wq[i]; wks[i] = g_wk[i];
                                          wvs[i] = g_wv[i]; }
  for (int i = tid; i < 12288; i += 256) { wihs[i] = g_wih[i]; whhs[i] = g_whh[i]; }
  for (int i = tid; i < 16384; i += 256) { sw1s[i] = g_sw1T[i]; sw2s[i] = g_sw2T[i]; }
  for (int i = tid; i < 512;  i += 256) slotsf[i] = g_prev[(size_t)b * 512 + i];
  for (int i = tid; i < 1024; i += 256) attnb[i] = (bf16_t)0.0f;
  for (int i = tid + 512; i < 1024; i += 256) { slotsb[i] = (bf16_t)0.0f; hbs[i] = (bf16_t)0.0f; }
  __syncthreads();

  for (int t = 0; t < Tn; ++t) {
    const float* lat = g_lat + ((size_t)b * Tn + t) * 512;
    for (int i = tid; i < 512; i += 256) latf[i] = lat[i];
    for (int i = tid; i < 512; i += 256) {
      const float v = slotsf[i] * 0.95f;
      slotsf[i] = v;
      slotsb[i] = (bf16_t)v;
    }
    __syncthreads();
    // ---- project tokens ----
    for (int e = tid; e < 4096; e += 256) {
      const int n = e >> 6, d = e & 63;
      float acc = pjbf[d] + posf[e];
#pragma unroll
      for (int c = 0; c < 8; ++c) acc += latf[c * 64 + n] * projw[c * 64 + d];
      tokf[e] = acc;
    }
    __syncthreads();
    // ---- LN per token ----
#pragma unroll
    for (int rr = 0; rr < 8; ++rr) {
      const int n = wave * 8 + rr;
      const float v0 = tokf[n * 64 + lane];
      const float v1 = tokf[n * 64 + 32 + lane];
      float s = v0 + v1, ss = v0 * v0 + v1 * v1;
#pragma unroll
      for (int off = 16; off > 0; off >>= 1) {
        s += __shfl_xor(s, off, 32);
        ss += __shfl_xor(ss, off, 32);
      }
      const float mean = s * 0.015625f;
      const float inv = rsqrtf(ss * 0.015625f - mean * mean + 1e-5f);
      tok[n * 64 + lane]      = (bf16_t)((v0 - mean) * inv * angf[lane] + anbf[lane]);
      tok[n * 64 + 32 + lane] = (bf16_t)((v1 - mean) * inv * angf[32 + lane] + anbf[32 + lane]);
    }
    __syncthreads();
    // ---- k (waves 0-3 -> kk) & v (waves 4-7 -> vT) ----
    {
      const int w4 = wave & 3;
      const bf16_t* wmat = (wave < 4) ? wks : wvs;
      v8f acc4[4] = {};
      const v16bf a0 = frag_a_ld(tok + w4 * 16 * 64, 64);
      const v16bf a1 = frag_a_ld(tok + w4 * 16 * 64 + 32, 64);
#pragma unroll
      for (int nb = 0; nb < 4; ++nb) {
        acc4[nb] = wmma_bf16(a0, frag_b_ld(wmat + nb * 16 * 64, 64), acc4[nb]);
        acc4[nb] = wmma_bf16(a1, frag_b_ld(wmat + nb * 16 * 64 + 32, 64), acc4[nb]);
      }
      const int nbase = w4 * 16 + ((lane >> 4) << 3);
#pragma unroll
      for (int nb = 0; nb < 4; ++nb) {
        const int d = nb * 16 + (lane & 15);
        if (wave < 4) {
#pragma unroll
          for (int r = 0; r < 8; ++r) kk[(nbase + r) * 64 + d] = (bf16_t)acc4[nb][r];
        } else {
          v8bf pack;
#pragma unroll
          for (int r = 0; r < 8; ++r) pack[r] = (bf16_t)acc4[nb][r];
          *reinterpret_cast<v8bf*>(vT + d * 64 + nbase) = pack;
        }
      }
    }
    __syncthreads();
    // ---- 3 slot-attention iterations ----
    for (int it = 0; it < 3; ++it) {
      if (wave < 4) {  // q = slots @ wq^T
        v8f acc = {};
        acc = wmma_bf16(frag_a_ld(slotsb, 64), frag_b_ld(wqs + wave * 16 * 64, 64), acc);
        acc = wmma_bf16(frag_a_ld(slotsb + 32, 64), frag_b_ld(wqs + wave * 16 * 64 + 32, 64), acc);
#pragma unroll
        for (int r = 0; r < 8; ++r) {
          const int m = r + ((lane >> 4) << 3);
          qb[m * 64 + wave * 16 + (lane & 15)] = (bf16_t)acc[r];
        }
      }
      __syncthreads();
      if (wave < 4) {  // attn logits = (q @ k^T) / 8
        v8f acc = {};
        acc = wmma_bf16(frag_a_ld(qb, 64), frag_b_ld(kk + wave * 16 * 64, 64), acc);
        acc = wmma_bf16(frag_a_ld(qb + 32, 64), frag_b_ld(kk + wave * 16 * 64 + 32, 64), acc);
#pragma unroll
        for (int r = 0; r < 8; ++r) {
          const int m = r + ((lane >> 4) << 3);
          if (m < 8) attnf[m * 64 + wave * 16 + (lane & 15)] = acc[r] * 0.125f;
        }
      }
      __syncthreads();
      if (tid < 64) {  // softmax over slots per token column
        float vals[8];
        float mx = -1e30f;
#pragma unroll
        for (int k8 = 0; k8 < 8; ++k8) { vals[k8] = attnf[k8 * 64 + tid]; mx = fmaxf(mx, vals[k8]); }
        float sum = 0.f;
#pragma unroll
        for (int k8 = 0; k8 < 8; ++k8) { vals[k8] = __expf(vals[k8] - mx); sum += vals[k8]; }
        const float inv = 1.f / sum;
#pragma unroll
        for (int k8 = 0; k8 < 8; ++k8) attnb[k8 * 64 + tid] = (bf16_t)(vals[k8] * inv);
      }
      __syncthreads();
      if (wave < 4) {  // upd = attn @ v
        v8f acc = {};
        acc = wmma_bf16(frag_a_ld(attnb, 64), frag_b_ld(vT + wave * 16 * 64, 64), acc);
        acc = wmma_bf16(frag_a_ld(attnb + 32, 64), frag_b_ld(vT + wave * 16 * 64 + 32, 64), acc);
#pragma unroll
        for (int r = 0; r < 8; ++r) {
          const int m = r + ((lane >> 4) << 3);
          updb[m * 64 + wave * 16 + (lane & 15)] = (bf16_t)acc[r];
        }
      }
      __syncthreads();
      // GRU gates: 24 N-tiles across 8 waves
      for (int j = wave; j < 24; j += 8) {
        const bool isX = (j < 12);
        const int nb = isX ? j : (j - 12);
        const bf16_t* W = isX ? wihs : whhs;
        const bf16_t* Af = isX ? updb : slotsb;
        v8f acc = {};
        acc = wmma_bf16(frag_a_ld(Af, 64), frag_b_ld(W + nb * 16 * 64, 64), acc);
        acc = wmma_bf16(frag_a_ld(Af + 32, 64), frag_b_ld(W + nb * 16 * 64 + 32, 64), acc);
        float* G = isX ? gxf : ghf;
        const float* Bv = isX ? bihf : bhhf;
#pragma unroll
        for (int r = 0; r < 8; ++r) {
          const int m = r + ((lane >> 4) << 3);
          const int c = nb * 16 + (lane & 15);
          if (m < 8) G[m * 192 + c] = acc[r] + Bv[c];
        }
      }
      __syncthreads();
      // GRU combine (gate order r,z,n)
      for (int e = tid; e < 512; e += 256) {
        const int k8 = e >> 6, d = e & 63;
        const float r  = sigmoid_f(gxf[k8 * 192 + d] + ghf[k8 * 192 + d]);
        const float z  = sigmoid_f(gxf[k8 * 192 + 64 + d] + ghf[k8 * 192 + 64 + d]);
        const float nn = tanhf(gxf[k8 * 192 + 128 + d] + r * ghf[k8 * 192 + 128 + d]);
        const float out = (1.f - z) * nn + z * slotsf[e];
        slotsf[e] = out;
        slotsb[e] = (bf16_t)out;
      }
      __syncthreads();
      {  // slot MLP: LN(slots), one row per wave
        const int k8 = wave;
        const float v0 = slotsf[k8 * 64 + lane];
        const float v1 = slotsf[k8 * 64 + 32 + lane];
        float s = v0 + v1, ss = v0 * v0 + v1 * v1;
#pragma unroll
        for (int off = 16; off > 0; off >>= 1) {
          s += __shfl_xor(s, off, 32);
          ss += __shfl_xor(ss, off, 32);
        }
        const float mean = s * 0.015625f;
        const float inv = rsqrtf(ss * 0.015625f - mean * mean + 1e-5f);
        hbs[k8 * 64 + lane]      = (bf16_t)((v0 - mean) * inv * sngf[lane] + snbf[lane]);
        hbs[k8 * 64 + 32 + lane] = (bf16_t)((v1 - mean) * inv * sngf[32 + lane] + snbf[32 + lane]);
      }
      __syncthreads();
      // hid = gelu(h @ smlp_w1 + b1)
      for (int nb = wave; nb < 16; nb += 8) {
        v8f acc = {};
        acc = wmma_bf16(frag_a_ld(hbs, 64), frag_b_ld(sw1s + nb * 16 * 64, 64), acc);
        acc = wmma_bf16(frag_a_ld(hbs + 32, 64), frag_b_ld(sw1s + nb * 16 * 64 + 32, 64), acc);
#pragma unroll
        for (int r = 0; r < 8; ++r) {
          const int m = r + ((lane >> 4) << 3);
          const int c = nb * 16 + (lane & 15);
          hidb[m * 256 + c] = (bf16_t)gelu_f(acc[r] + sb1f[c]);
        }
      }
      __syncthreads();
      if (wave < 4) {  // slots += hid @ smlp_w2 + b2
        v8f acc = {};
#pragma unroll
        for (int kc = 0; kc < 8; ++kc) {
          acc = wmma_bf16(frag_a_ld(hidb + kc * 32, 256),
                          frag_b_ld(sw2s + wave * 16 * 256 + kc * 32, 256), acc);
        }
#pragma unroll
        for (int r = 0; r < 8; ++r) {
          const int m = r + ((lane >> 4) << 3);
          const int d = wave * 16 + (lane & 15);
          if (m < 8) {
            const float out = slotsf[m * 64 + d] + acc[r] + sb2f[d];
            slotsf[m * 64 + d] = out;
            slotsb[m * 64 + d] = (bf16_t)out;
          }
        }
      }
      __syncthreads();
    }
    float* outp = all_slots + ((size_t)b * Tn + t) * 512;
    for (int i = tid; i < 512; i += 256) outp[i] = slotsf[i];
    if (t == Tn - 1) {
      float* fs = final_slots + (size_t)b * 512;
      for (int i = tid; i < 512; i += 256) fs[i] = slotsf[i];
    }
    __syncthreads();
  }
}

// ===========================================================================
// Host driver
// ===========================================================================
extern "C" void kernel_launch(void* const* d_in, const int* in_sizes, int n_in,
                              void* d_out, int out_size, void* d_ws, size_t ws_size,
                              hipStream_t stream) {
  (void)in_sizes; (void)n_in; (void)out_size; (void)ws_size;
  const float* latents    = (const float*)d_in[0];
  const float* actions    = (const float*)d_in[1];
  const float* prev_slots = (const float*)d_in[2];
  const float* proj_w = (const float*)d_in[3];
  const float* proj_b = (const float*)d_in[4];
  const float* pos    = (const float*)d_in[5];
  const float* an_g   = (const float*)d_in[6];
  const float* an_b   = (const float*)d_in[7];
  const float* wq     = (const float*)d_in[8];
  const float* wk     = (const float*)d_in[9];
  const float* wvp    = (const float*)d_in[10];
  const float* w_ih   = (const float*)d_in[11];
  const float* w_hh   = (const float*)d_in[12];
  const float* b_ih   = (const float*)d_in[13];
  const float* b_hh   = (const float*)d_in[14];
  const float* sn_g   = (const float*)d_in[15];
  const float* sn_b   = (const float*)d_in[16];
  const float* smlp_w1 = (const float*)d_in[17];
  const float* smlp_b1 = (const float*)d_in[18];
  const float* smlp_w2 = (const float*)d_in[19];
  const float* smlp_b2 = (const float*)d_in[20];
  const float* act_w  = (const float*)d_in[21];
  const float* act_b  = (const float*)d_in[22];
  const float* dec_w1 = (const float*)d_in[23 + 12 * NL + 0];
  const float* dec_b1 = (const float*)d_in[23 + 12 * NL + 1];
  const float* dec_w2 = (const float*)d_in[23 + 12 * NL + 2];
  const float* dec_b2 = (const float*)d_in[23 + 12 * NL + 3];
  const float* on_g   = (const float*)d_in[23 + 12 * NL + 4];
  const float* on_b   = (const float*)d_in[23 + 12 * NL + 5];
  const float* scale  = (const float*)d_in[23 + 12 * NL + 6];

  size_t off = 0;
  auto alloc = [&](size_t bytes) -> char* {
    off = (off + 255) & ~(size_t)255;
    char* p = (char*)d_ws + off;
    off += bytes;
    return p;
  };
  bf16_t* actions_bf = (bf16_t*)alloc((size_t)NTOK * AKP * 2);  // padded [NTOK,32]
  bf16_t* wqW  = (bf16_t*)alloc(4096 * 2);
  bf16_t* wkW  = (bf16_t*)alloc(4096 * 2);
  bf16_t* wvW  = (bf16_t*)alloc(4096 * 2);
  bf16_t* wihW = (bf16_t*)alloc(12288 * 2);
  bf16_t* whhW = (bf16_t*)alloc(12288 * 2);
  bf16_t* sw1T = (bf16_t*)alloc(16384 * 2);
  bf16_t* sw2T = (bf16_t*)alloc(16384 * 2);
  bf16_t* actwT = (bf16_t*)alloc((size_t)DM * AKP * 2);         // padded [512,32]
  bf16_t* BmT[NL]; bf16_t* CmT[NL]; bf16_t* w1T[NL]; bf16_t* w2T[NL];
  for (int l = 0; l < NL; ++l) {
    BmT[l] = (bf16_t*)alloc((size_t)DM * DS * 2);
    CmT[l] = (bf16_t*)alloc((size_t)DS * DM * 2);
    w1T[l] = (bf16_t*)alloc((size_t)DM * MH * 2);
    w2T[l] = (bf16_t*)alloc((size_t)MH * DM * 2);
  }
  bf16_t* dw1T = (bf16_t*)alloc((size_t)DM * HID * 2);
  bf16_t* dw2T = (bf16_t*)alloc((size_t)HID * LAT * 2);
  float*  all_slots = (float*)alloc((size_t)NTOK * DM * 4);
  float*  x   = (float*)alloc((size_t)NTOK * DM * 4);
  bf16_t* xb  = (bf16_t*)alloc((size_t)NTOK * DM * 2);
  bf16_t* hb  = (bf16_t*)alloc((size_t)NTOK * DM * 2);
  bf16_t* hid = (bf16_t*)alloc((size_t)NTOK * MH * 2);
  float*  u   = (float*)alloc((size_t)NTOK * DS * 4);
  bf16_t* ys  = (bf16_t*)alloc((size_t)NTOK * DS * 2);
  bf16_t* d1  = (bf16_t*)alloc((size_t)NTOK * HID * 2);
  float*  d2  = (float*)alloc((size_t)NTOK * LAT * 4);

  float* out_pred  = (float*)d_out;                 // [B,T,512]
  float* out_slots = out_pred + (size_t)NTOK * DM;  // [B,8,64]

  auto cvt = [&](const float* s, bf16_t* d, int r, int c, int tr) {
    const int total = r * c;
    cvt_kernel<<<(total + 255) / 256, 256, 0, stream>>>(s, d, r, c, tr);
  };
  auto gemm = [&](const bf16_t* A, const bf16_t* Bw, const float* bias,
                  const float* resid, float* Cf, bf16_t* Cb, int M, int N, int K,
                  int gl) {
    dim3 grid((N + GTN - 1) / GTN, (M + GTM - 1) / GTM);
    gemm_bf16_kernel<<<grid, 256, 0, stream>>>(A, Bw, bias, resid, Cf, Cb, M, N, K, gl);
  };
  auto ln = [&](const float* xi, const float* g, const float* bv, bf16_t* ob,
                float* of, const float* sc, int D, int rows) {
    ln_kernel<<<rows, 256, 0, stream>>>(xi, g, bv, ob, of, sc, D);
  };

  // ---- 1) weight/activation conversions (GEMM operands stored [N,K]) ----
  cvt_pad_kernel<<<(NTOK * AKP + 255) / 256, 256, 0, stream>>>(
      actions, actions_bf, NTOK, 18, AKP, 0);              // [NTOK,32] zero-padded
  cvt_pad_kernel<<<(DM * AKP + 255) / 256, 256, 0, stream>>>(
      act_w, actwT, 18, DM, AKP, 1);                       // [512,32] zero-padded
  cvt(wq, wqW, 64, 64, 0);
  cvt(wk, wkW, 64, 64, 0);
  cvt(wvp, wvW, 64, 64, 0);
  cvt(w_ih, wihW, 192, 64, 0);
  cvt(w_hh, whhW, 192, 64, 0);
  cvt(smlp_w1, sw1T, 64, 256, 1);
  cvt(smlp_w2, sw2T, 256, 64, 1);
  for (int l = 0; l < NL; ++l) {
    const int base = 23 + 12 * l;
    cvt((const float*)d_in[base + 2], BmT[l], DM, DS, 1);   // -> [256,512]
    cvt((const float*)d_in[base + 3], CmT[l], DS, DM, 1);   // -> [512,256]
    cvt((const float*)d_in[base + 8], w1T[l], DM, MH, 1);   // -> [2048,512]
    cvt((const float*)d_in[base + 10], w2T[l], MH, DM, 1);  // -> [512,2048]
  }
  cvt(dec_w1, dw1T, DM, HID, 1);
  cvt(dec_w2, dw2T, HID, LAT, 1);

  // ---- 2) slot-attention scan ----
  (void)hipFuncSetAttribute(reinterpret_cast<const void*>(slot_attn_kernel),
                            hipFuncAttributeMaxDynamicSharedMemorySize,
                            SLOT_LDS_BYTES);
  slot_attn_kernel<<<MB, 256, SLOT_LDS_BYTES, stream>>>(
      latents, prev_slots, proj_w, proj_b, pos, an_g, an_b, wqW, wkW, wvW, wihW,
      whhW, b_ih, b_hh, sn_g, sn_b, sw1T, smlp_b1, sw2T, smlp_b2, all_slots,
      out_slots, MT);

  // ---- 3) action conditioning: x = slots + actions @ act_w + act_b ----
  gemm(actions_bf, actwT, act_b, all_slots, x, nullptr, NTOK, DM, AKP, 0);

  // ---- 4) BSD trunk ----
  for (int l = 0; l < NL; ++l) {
    const int base = 23 + 12 * l;
    const float* ln1_g = (const float*)d_in[base + 0];
    const float* ln1_b = (const float*)d_in[base + 1];
    const float* lam   = (const float*)d_in[base + 4];
    const float* theta = (const float*)d_in[base + 5];
    const float* ln2_g = (const float*)d_in[base + 6];
    const float* ln2_b = (const float*)d_in[base + 7];
    const float* mb1   = (const float*)d_in[base + 9];
    const float* mb2   = (const float*)d_in[base + 11];

    ln(x, ln1_g, ln1_b, hb, nullptr, nullptr, DM, NTOK);
    gemm(hb, BmT[l], nullptr, nullptr, u, nullptr, NTOK, DS, DM, 0);
    ssm_scan_kernel<<<(MB * DS + 255) / 256, 256, 0, stream>>>(u, lam, theta, ys,
                                                               MB, MT, DS);
    gemm(ys, CmT[l], nullptr, x, x, nullptr, NTOK, DM, DS, 0);
    ln(x, ln2_g, ln2_b, hb, nullptr, nullptr, DM, NTOK);
    gemm(hb, w1T[l], mb1, nullptr, nullptr, hid, NTOK, MH, DM, 1);
    gemm(hid, w2T[l], mb2, x, x, (l == NL - 1) ? xb : nullptr, NTOK, DM, MH, 0);
  }

  // ---- 5) decoder + output LN*scale ----
  gemm(xb, dw1T, dec_b1, nullptr, nullptr, d1, NTOK, HID, DM, 1);
  gemm(d1, dw2T, dec_b2, nullptr, d2, nullptr, NTOK, LAT, HID, 0);
  ln(d2, on_g, on_b, nullptr, out_pred, scale, LAT, NTOK);
}
